// HyperGraphEncoderLayer_8529805050069
// MI455X (gfx1250) — matrix operand order repaired and checked by reading, not compiled, not run on hardware
//
#include <hip/hip_runtime.h>
#include <hip/hip_bf16.h>

#define NN   100000
#define EE   20000
#define DD   64
#define HH   4
#define NNZ_ 500000

typedef __attribute__((ext_vector_type(16))) _Float16 v16h;
typedef __attribute__((ext_vector_type(8)))  _Float16 v8h;
typedef __attribute__((ext_vector_type(8)))  float    v8f;

// ---------------- utility kernels ----------------

__global__ __launch_bounds__(256) void k_zero(float* p, int n) {
  int i = blockIdx.x * 256 + threadIdx.x;
  if (i < n) p[i] = 0.f;
}

__global__ __launch_bounds__(256) void k_fill_i32(int* p, int n, int v) {
  int i = blockIdx.x * 256 + threadIdx.x;
  if (i < n) p[i] = v;
}

__global__ __launch_bounds__(256) void k_cvt_f16(const float* __restrict__ s, _Float16* __restrict__ d, int n) {
  int i = blockIdx.x * 256 + threadIdx.x;
  if (i < n) d[i] = (_Float16)s[i];
}

// monotone int key for f32 atomic max
__device__ __forceinline__ int f2key(float x) { int k = __float_as_int(x); return k < 0 ? (k ^ 0x7fffffff) : k; }
__device__ __forceinline__ float key2f(int k) { return __int_as_float(k < 0 ? (k ^ 0x7fffffff) : k); }

// ---------------- stage 1: hyperedge features omega ----------------
// one wave per nnz entry; lanes cover the 64 feature dims in 2 steps
__global__ __launch_bounds__(256) void k_scatter_omega(const float* __restrict__ x,
    const int* __restrict__ row, const int* __restrict__ col, const float* __restrict__ val,
    float* colsum, float* omega, float* bdeg, float* ddeg) {
  int wid  = (blockIdx.x * 256 + threadIdx.x) >> 5;
  int lane = threadIdx.x & 31;
  if (wid >= NNZ_) return;
  int r = row[wid], c = col[wid];
  float v = val[wid];
  // prefetch the x-row a future wave will gather (spread lanes across the 256B row)
  if (wid + 32 < NNZ_) {
    int rpf = row[wid + 32];
    __builtin_prefetch((const char*)(x + (size_t)rpf * DD) + lane * 8, 0, 1);
  }
  if (lane == 0) {
    atomicAdd(&colsum[c], v);
    atomicAdd(&bdeg[c], 1.0f);
    atomicAdd(&ddeg[r], val[c]);        // segment_sum(adj_val[adj_col], adj_row)
  }
  const float* xr = x + (size_t)r * DD;
  float* om = omega + (size_t)c * DD;
  atomicAdd(&om[lane],      v * xr[lane]);
  atomicAdd(&om[lane + 32], v * xr[lane + 32]);
}

__global__ __launch_bounds__(256) void k_omega_norm(const float* __restrict__ omega,
    const float* __restrict__ colsum, _Float16* __restrict__ oh, int n) {
  int i = blockIdx.x * 256 + threadIdx.x;
  if (i >= n) return;
  float cs = colsum[i >> 6];
  oh[i] = (_Float16)(cs > 0.f ? omega[i] / cs : 0.f);
}

// ---------------- WMMA GEMM:  out[M,NCOLS] = A[M,64] @ W[NCOLS,64]^T ----------------
// one wave computes a 16x64 output span: A fragments loaded once, 4 j-tiles,
// 8x v_wmma_f32_16x16x32_f16. NCOLS is compile-time so epilogue stores fold
// the row stride into immediate offsets off a single base address.
template <int NCOLS, bool OUT16, bool HASBIAS, bool LEAKY>
__global__ __launch_bounds__(256) void k_gemm_f16_wmma(const _Float16* __restrict__ A,
    const _Float16* __restrict__ W, int Mrows,
    void* __restrict__ outp, const float* __restrict__ bias, float negSlope) {
  int wid  = (blockIdx.x * 256 + threadIdx.x) >> 5;
  int lane = threadIdx.x & 31;
  constexpr int spansX = NCOLS >> 6;       // 64-column spans
  int nspans = (Mrows >> 4) * spansX;
  if (wid >= nspans) return;               // wave-uniform: EXEC all-ones past here
  int ti = wid / spansX;
  int jspan = (wid - ti * spansX) << 6;
  int row0 = ti << 4;
  int m  = lane & 15;
  int hf = lane >> 4;

  // A fragment: lane holds row m; K runs
  // {hf*16..hf*16+15} (matches B) -> contiguous 16B loads
  const _Float16* Ar = A + (size_t)(row0 + m) * DD + hf * 16;
  v8h a00 = *(const v8h*)(Ar);
  v8h a01 = *(const v8h*)(Ar + 8);
  v8h a10 = *(const v8h*)(Ar + 32);
  v8h a11 = *(const v8h*)(Ar + 40);
  v16h a0, a1;
  #pragma unroll
  for (int t = 0; t < 8; ++t) { a0[t] = a00[t]; a0[t + 8] = a01[t]; a1[t] = a10[t]; a1[t + 8] = a11[t]; }

  #pragma unroll
  for (int jt = 0; jt < 4; ++jt) {
    int j0 = jspan + jt * 16;
    // B fragment: lane holds column j0+m of W^T == row j0+m of W; K = 16*hf + p (+32*kb)
    const _Float16* Wr = W + (size_t)(j0 + m) * DD + hf * 16;
    v16h b0 = *(const v16h*)(Wr);
    v16h b1 = *(const v16h*)(Wr + 32);
    v8f c = {};
    c = __builtin_amdgcn_wmma_f32_16x16x32_f16(false, a0, false, b0, (short)0, c, false, false);
    c = __builtin_amdgcn_wmma_f32_16x16x32_f16(false, a1, false, b1, (short)0, c, false, false);

    int col = j0 + m;
    float bv = 0.f;
    if (HASBIAS) bv = bias[col & 63];
    size_t base = (size_t)(row0 + hf * 8) * NCOLS + col;   // C/D layout: M = r2 + 8*hf
    if (OUT16) {
      _Float16* op = (_Float16*)outp + base;
      #pragma unroll
      for (int r2 = 0; r2 < 8; ++r2) {
        float vv = c[r2];
        if (HASBIAS) vv += bv;
        if (LEAKY) vv = vv > 0.f ? vv : vv * negSlope;
        op[(size_t)r2 * NCOLS] = (_Float16)vv;             // constant-offset stores
      }
    } else {
      float* op = (float*)outp + base;
      #pragma unroll
      for (int r2 = 0; r2 < 8; ++r2) {
        float vv = c[r2];
        if (HASBIAS) vv += bv;
        if (LEAKY) vv = vv > 0.f ? vv : vv * negSlope;
        op[(size_t)r2 * NCOLS] = vv;
      }
    }
  }
}

// ---------------- attention scores: s[n,h] = sum_d feat[n,h,d]*att[h,d] ----------------
__global__ __launch_bounds__(256) void k_att_dot(const _Float16* __restrict__ Mh,
    const float* __restrict__ att, int rows, int attOff, float* __restrict__ sOut) {
  int wid  = (blockIdx.x * 256 + threadIdx.x) >> 5;
  int lane = threadIdx.x & 31;
  if (wid >= rows) return;
  v8h mv = *(const v8h*)(Mh + (size_t)wid * (HH * DD) + lane * 8);
  int h = lane >> 3;
  const float* ap = att + h * (2 * DD) + attOff + (lane & 7) * 8;
  float acc = 0.f;
  #pragma unroll
  for (int t = 0; t < 8; ++t) acc += (float)mv[t] * ap[t];
  acc += __shfl_xor(acc, 1, 32);
  acc += __shfl_xor(acc, 2, 32);
  acc += __shfl_xor(acc, 4, 32);
  if ((lane & 7) == 0) sOut[(size_t)wid * HH + h] = acc;
}

// ---------------- softmax over hyperedge members ----------------
__global__ __launch_bounds__(256) void k_alpha_max(const int* __restrict__ row, const int* __restrict__ col,
    const float* __restrict__ s_n, const float* __restrict__ s_e, int* amax) {
  int i = blockIdx.x * 256 + threadIdx.x;
  if (i >= NNZ_) return;
  int r = row[i], c = col[i];
  #pragma unroll
  for (int h = 0; h < HH; ++h) {
    float a = s_n[(size_t)r * HH + h] + s_e[(size_t)c * HH + h];
    a = a > 0.f ? a : 0.2f * a;
    atomicMax(&amax[(size_t)c * HH + h], f2key(a));
  }
}

__global__ __launch_bounds__(256) void k_alpha_exp(const int* __restrict__ row, const int* __restrict__ col,
    const float* __restrict__ s_n, const float* __restrict__ s_e, const int* __restrict__ amax,
    float* __restrict__ ea, float* esum) {
  int i = blockIdx.x * 256 + threadIdx.x;
  if (i >= NNZ_) return;
  int r = row[i], c = col[i];
  #pragma unroll
  for (int h = 0; h < HH; ++h) {
    float a = s_n[(size_t)r * HH + h] + s_e[(size_t)c * HH + h];
    a = a > 0.f ? a : 0.2f * a;
    float e = __expf(a - key2f(amax[(size_t)c * HH + h]));
    ea[(size_t)i * HH + h] = e;
    atomicAdd(&esum[(size_t)c * HH + h], e);
  }
}

// ---------------- message passing ----------------
// msg1: edge_out[c,h,d] += b_inv[c]*alpha[i,h]*xh[r,h,d]; lane owns 8 contiguous dims (one head)
__global__ __launch_bounds__(256) void k_msg1(const int* __restrict__ row, const int* __restrict__ col,
    const _Float16* __restrict__ xh, const float* __restrict__ ea, const float* __restrict__ esum,
    const float* __restrict__ bdeg, float* __restrict__ eout) {
  int wid  = (blockIdx.x * 256 + threadIdx.x) >> 5;
  int lane = threadIdx.x & 31;
  if (wid >= NNZ_) return;
  int r = row[wid], c = col[wid];
  // prefetch the gathered xh row (512B) of a future entry; lanes cover it 16B apart
  if (wid + 32 < NNZ_) {
    int rpf = row[wid + 32];
    __builtin_prefetch((const char*)(xh + (size_t)rpf * (HH * DD)) + lane * 16, 0, 1);
  }
  float bd = bdeg[c];
  float binv = bd > 0.f ? 1.f / bd : 0.f;
  int h = lane >> 3;
  float es = esum[(size_t)c * HH + h];
  float al = es > 0.f ? ea[(size_t)wid * HH + h] / es : 0.f;
  float w = binv * al;
  v8h xv = *(const v8h*)(xh + (size_t)r * (HH * DD) + lane * 8);
  float* eo = eout + (size_t)c * (HH * DD) + lane * 8;
  #pragma unroll
  for (int t = 0; t < 8; ++t) atomicAdd(&eo[t], w * (float)xv[t]);
}

// msg2 with head-mean folded in: node_conv[r,d] += 0.25*d_inv[r]*sum_h alpha[i,h]*edge_out[c,h,d]
__global__ __launch_bounds__(256) void k_msg2(const int* __restrict__ row, const int* __restrict__ col,
    const float* __restrict__ ea, const float* __restrict__ esum, const float* __restrict__ ddeg,
    const float* __restrict__ eout, float* __restrict__ nconv) {
  int wid  = (blockIdx.x * 256 + threadIdx.x) >> 5;
  int lane = threadIdx.x & 31;
  if (wid >= NNZ_) return;
  int r = row[wid], c = col[wid];
  // prefetch the gathered edge_out row (1KB) of a future entry; lanes cover it 32B apart
  if (wid + 32 < NNZ_) {
    int cpf = col[wid + 32];
    __builtin_prefetch((const char*)(eout + (size_t)cpf * (HH * DD)) + lane * 32, 0, 1);
  }
  float dd = ddeg[r];
  float dinv = dd > 0.f ? 0.25f / dd : 0.f;
  float al[HH];
  #pragma unroll
  for (int h = 0; h < HH; ++h) {
    float es = esum[(size_t)c * HH + h];
    al[h] = es > 0.f ? ea[(size_t)wid * HH + h] / es : 0.f;
  }
  const float* eo = eout + (size_t)c * (HH * DD);
  #pragma unroll
  for (int s = 0; s < 2; ++s) {
    int d = lane + s * 32;
    float v = al[0] * eo[d] + al[1] * eo[DD + d] + al[2] * eo[2 * DD + d] + al[3] * eo[3 * DD + d];
    atomicAdd(&nconv[(size_t)r * DD + d], dinv * v);
  }
}

// ---------------- residual + LayerNorm (wave32 per row) ----------------
template <bool HASBIAS, bool OUT16>
__global__ __launch_bounds__(256) void k_add_ln(const float* __restrict__ a,
    const float* __restrict__ bsrc, const float* __restrict__ bias,
    const float* __restrict__ g, const float* __restrict__ bb,
    float* __restrict__ out32, _Float16* out16) {
  int wid  = (blockIdx.x * 256 + threadIdx.x) >> 5;
  int lane = threadIdx.x & 31;
  if (wid >= NN) return;
  size_t off = (size_t)wid * DD;
  int d0 = lane, d1 = lane + 32;
  float t0 = a[off + d0] + bsrc[off + d0];
  float t1 = a[off + d1] + bsrc[off + d1];
  if (HASBIAS) { t0 += bias[d0]; t1 += bias[d1]; }
  float s = t0 + t1;
  #pragma unroll
  for (int mk = 1; mk < 32; mk <<= 1) s += __shfl_xor(s, mk, 32);
  float mu = s * (1.f / 64.f);
  float e0 = t0 - mu, e1 = t1 - mu;
  float vq = e0 * e0 + e1 * e1;
  #pragma unroll
  for (int mk = 1; mk < 32; mk <<= 1) vq += __shfl_xor(vq, mk, 32);
  float inv = rsqrtf(vq * (1.f / 64.f) + 1e-5f);
  float o0 = e0 * inv * g[d0] + bb[d0];
  float o1 = e1 * inv * g[d1] + bb[d1];
  out32[off + d0] = o0; out32[off + d1] = o1;
  if (OUT16) { out16[off + d0] = (_Float16)o0; out16[off + d1] = (_Float16)o1; }
}

// ---------------- host ----------------
extern "C" void kernel_launch(void* const* d_in, const int* in_sizes, int n_in,
                              void* d_out, int out_size, void* d_ws, size_t ws_size,
                              hipStream_t stream) {
  (void)in_sizes; (void)n_in; (void)out_size; (void)ws_size;
  const float* x       = (const float*)d_in[0];
  const int*   adj_row = (const int*)d_in[1];
  const int*   adj_col = (const int*)d_in[2];
  const float* adj_val = (const float*)d_in[3];
  const float* lin_w   = (const float*)d_in[5];
  const float* att     = (const float*)d_in[6];
  const float* bias    = (const float*)d_in[7];
  const float* ln1_g   = (const float*)d_in[8];
  const float* ln1_b   = (const float*)d_in[9];
  const float* ln2_g   = (const float*)d_in[10];
  const float* ln2_b   = (const float*)d_in[11];
  const float* w1      = (const float*)d_in[12];
  const float* b1      = (const float*)d_in[13];
  const float* w2      = (const float*)d_in[14];
  const float* b2      = (const float*)d_in[15];

  char* ws = (char*)d_ws;
  size_t o = 0;
  auto al = [&](size_t bytes) { size_t r = o; o += (bytes + 255) & ~(size_t)255; return r; };
  // accumulators (zeroed each call), contiguous
  size_t off_colsum = al((size_t)EE * 4);
  size_t off_omega  = al((size_t)EE * DD * 4);
  size_t off_esum   = al((size_t)EE * HH * 4);
  size_t off_bdeg   = al((size_t)EE * 4);
  size_t off_ddeg   = al((size_t)NN * 4);
  size_t off_eout   = al((size_t)EE * HH * DD * 4);
  size_t off_nconv  = al((size_t)NN * DD * 4);
  size_t zeroBytes  = o;
  // non-zeroed scratch
  size_t off_amax   = al((size_t)EE * HH * 4);
  size_t off_x16    = al((size_t)NN * DD * 2);
  size_t off_wh     = al((size_t)HH * DD * DD * 2);
  size_t off_w1h    = al((size_t)DD * DD * 2);
  size_t off_w2h    = al((size_t)DD * DD * 2);
  size_t off_oh     = al((size_t)EE * DD * 2);
  size_t off_xhh    = al((size_t)NN * HH * DD * 2);
  size_t off_ehh    = al((size_t)EE * HH * DD * 2);
  size_t off_sn     = al((size_t)NN * HH * 4);
  size_t off_se     = al((size_t)EE * HH * 4);
  size_t off_ea     = al((size_t)NNZ_ * HH * 4);
  size_t off_hbuf   = al((size_t)NN * DD * 4);
  size_t off_hh     = al((size_t)NN * DD * 2);
  size_t off_t1     = al((size_t)NN * DD * 2);
  size_t off_ffn    = al((size_t)NN * DD * 4);

  float*    colsum = (float*)(ws + off_colsum);
  float*    omega  = (float*)(ws + off_omega);
  float*    esum   = (float*)(ws + off_esum);
  float*    bdeg   = (float*)(ws + off_bdeg);
  float*    ddeg   = (float*)(ws + off_ddeg);
  float*    eout   = (float*)(ws + off_eout);
  float*    nconv  = (float*)(ws + off_nconv);
  int*      amax   = (int*)(ws + off_amax);
  _Float16* x16    = (_Float16*)(ws + off_x16);
  _Float16* wh     = (_Float16*)(ws + off_wh);
  _Float16* w1h    = (_Float16*)(ws + off_w1h);
  _Float16* w2h    = (_Float16*)(ws + off_w2h);
  _Float16* oh     = (_Float16*)(ws + off_oh);
  _Float16* xhh    = (_Float16*)(ws + off_xhh);
  _Float16* ehh    = (_Float16*)(ws + off_ehh);
  float*    sn     = (float*)(ws + off_sn);
  float*    se     = (float*)(ws + off_se);
  float*    ea     = (float*)(ws + off_ea);
  float*    hbuf   = (float*)(ws + off_hbuf);
  _Float16* hh     = (_Float16*)(ws + off_hh);
  _Float16* t1h    = (_Float16*)(ws + off_t1);
  float*    ffn    = (float*)(ws + off_ffn);

  int zn = (int)(zeroBytes / 4);
  k_zero<<<(zn + 255) / 256, 256, 0, stream>>>((float*)ws, zn);
  k_fill_i32<<<(EE * HH + 255) / 256, 256, 0, stream>>>(amax, EE * HH, (int)0x80000000);

  k_cvt_f16<<<(NN * DD + 255) / 256, 256, 0, stream>>>(x, x16, NN * DD);
  k_cvt_f16<<<(HH * DD * DD + 255) / 256, 256, 0, stream>>>(lin_w, wh, HH * DD * DD);
  k_cvt_f16<<<(DD * DD + 255) / 256, 256, 0, stream>>>(w1, w1h, DD * DD);
  k_cvt_f16<<<(DD * DD + 255) / 256, 256, 0, stream>>>(w2, w2h, DD * DD);

  k_scatter_omega<<<NNZ_ / 8, 256, 0, stream>>>(x, adj_row, adj_col, adj_val, colsum, omega, bdeg, ddeg);
  k_omega_norm<<<EE * DD / 256, 256, 0, stream>>>(omega, colsum, oh, EE * DD);

  // xh = x @ lin_w^T (100000x256): 6250*4 spans; eh = omega @ lin_w^T (20000x256): 1250*4 spans
  k_gemm_f16_wmma<256, true, false, false><<<(NN / 16) * 4 / 8, 256, 0, stream>>>(x16, wh, NN, xhh, nullptr, 1.0f);
  k_gemm_f16_wmma<256, true, false, false><<<(EE / 16) * 4 / 8, 256, 0, stream>>>(oh, wh, EE, ehh, nullptr, 1.0f);

  k_att_dot<<<NN / 8, 256, 0, stream>>>(xhh, att, NN, 0, sn);
  k_att_dot<<<EE / 8, 256, 0, stream>>>(ehh, att, EE, DD, se);

  k_alpha_max<<<(NNZ_ + 255) / 256, 256, 0, stream>>>(adj_row, adj_col, sn, se, amax);
  k_alpha_exp<<<(NNZ_ + 255) / 256, 256, 0, stream>>>(adj_row, adj_col, sn, se, amax, ea, esum);

  k_msg1<<<NNZ_ / 8, 256, 0, stream>>>(adj_row, adj_col, xhh, ea, esum, bdeg, eout);
  k_msg2<<<NNZ_ / 8, 256, 0, stream>>>(adj_row, adj_col, ea, esum, ddeg, eout, nconv);

  // h = LN1(x + conv)
  k_add_ln<true, true><<<NN / 8, 256, 0, stream>>>(x, nconv, bias, ln1_g, ln1_b, hbuf, hh);

  // FFN: t1 = h @ w1^T + b1; ffn = leaky(t1 @ w2^T + b2, 0.01). 6250 spans each.
  k_gemm_f16_wmma<64, true, true, false><<<(NN / 16 + 7) / 8, 256, 0, stream>>>(hh, w1h, NN, t1h, b1, 1.0f);
  k_gemm_f16_wmma<64, false, true, true><<<(NN / 16 + 7) / 8, 256, 0, stream>>>(t1h, w2h, NN, ffn, b2, 0.01f);

  // out = LN2(h + ffn)
  k_add_ln<false, false><<<NN / 8, 256, 0, stream>>>(hbuf, ffn, nullptr, ln2_g, ln2_b, (float*)d_out, nullptr);
}
